// DecoderLayer1MoEOnly_54855322305271
// MI455X (gfx1250) — matrix operand order repaired and checked by reading, compiled, not run on hardware
//
#include <hip/hip_runtime.h>
#include <hip/hip_bf16.h>
#include <math.h>

// ---------------- problem constants (match reference) ----------------
#define T_TOK   1024
#define HDIM    1280
#define NEXP    64
#define KTOP    6
#define FDIM    896
#define FSDIM   1792
#define CAP     192
#define NFLAT   (T_TOK * KTOP)

// ---------------- WMMA types ----------------
typedef __attribute__((ext_vector_type(16))) __bf16 bf16x16;
typedef __attribute__((ext_vector_type(8)))  float  f32x8;

union FragCvt {
    struct { uint4 a, b; } u;
    bf16x16 v;
};

__device__ __forceinline__ unsigned short f2bf(float f) {
    unsigned u = __builtin_bit_cast(unsigned, f);
    unsigned r = u + 0x7FFFu + ((u >> 16) & 1u);   // round-to-nearest-even
    return (unsigned short)(r >> 16);
}

// packed f32 pair -> bf16x2: force exactly one v_cvt_pk_bf16_f32 with the
// pairing we need (lo -> [15:0], hi -> [31:16]); avoids compiler re-packing.
__device__ __forceinline__ unsigned cvt2(float lo, float hi) {
    unsigned d;
    asm("v_cvt_pk_bf16_f32 %0, %1, %2" : "=v"(d) : "v"(lo), "v"(hi));
    return d;
}

__device__ __forceinline__ f32x8 wmma_bf16(bf16x16 a, bf16x16 b, f32x8 c) {
    return __builtin_amdgcn_wmma_f32_16x16x32_bf16(
        false, a, false, b, (short)0, c, false, false);
}

// CDNA5 async global->LDS copy (16B per lane), tracked by ASYNCcnt
__device__ __forceinline__ void async_g2l_b128(unsigned lds_addr, const void* gaddr) {
    asm volatile("global_load_async_to_lds_b128 %0, %1, off"
                 :: "v"(lds_addr), "v"(gaddr) : "memory");
}
__device__ __forceinline__ void wait_async0() {
    asm volatile("s_wait_asynccnt 0" ::: "memory");
}
__device__ __forceinline__ unsigned lds_off32(const void* p) {
    return (unsigned)(unsigned long long)p;   // LDS aperture: addr[31:0] = LDS offset
}

// ---------------- kernel 1: RMSNorm ----------------
__global__ __launch_bounds__(256) void rmsnorm_kernel(
    const float* __restrict__ x, const float* __restrict__ w,
    float* __restrict__ xn, unsigned short* __restrict__ xnb)
{
    __shared__ float sm[256];
    const int t = blockIdx.x, tid = threadIdx.x;
    const float* xr = x + (long)t * HDIM;
    float s = 0.f;
    for (int h = tid; h < HDIM; h += 256) { float v = xr[h]; s += v * v; }
    sm[tid] = s;
    __syncthreads();
    for (int off = 128; off > 0; off >>= 1) {
        if (tid < off) sm[tid] += sm[tid + off];
        __syncthreads();
    }
    const float inv = rsqrtf(sm[0] / (float)HDIM + 1e-6f);
    for (int h = tid; h < HDIM; h += 256) {
        float v = xr[h] * inv * w[h];
        xn[(long)t * HDIM + h]  = v;
        xnb[(long)t * HDIM + h] = f2bf(v);
    }
}

// ---------------- kernel 2: router + softmax + top-6 ----------------
__global__ __launch_bounds__(64) void router_kernel(
    const float* __restrict__ xn, const float* __restrict__ rw,
    int* __restrict__ flat_e, float* __restrict__ flat_w)
{
    __shared__ float xs[HDIM];
    __shared__ float probs[NEXP];
    const int t = blockIdx.x, tid = threadIdx.x;
    for (int h = tid; h < HDIM; h += 64) xs[h] = xn[(long)t * HDIM + h];
    __syncthreads();
    float acc = 0.f;
    for (int h = 0; h < HDIM; ++h) acc = fmaf(xs[h], rw[(long)h * NEXP + tid], acc);
    probs[tid] = acc;
    __syncthreads();
    if (tid == 0) {
        float mx = probs[0];
        for (int e = 1; e < NEXP; ++e) mx = fmaxf(mx, probs[e]);
        float ssum = 0.f;
        for (int e = 0; e < NEXP; ++e) { float p = __expf(probs[e] - mx); probs[e] = p; ssum += p; }
        const float inv = 1.f / ssum;
        for (int k = 0; k < KTOP; ++k) {
            int best = 0; float bv = -1.f;
            for (int e = 0; e < NEXP; ++e)
                if (probs[e] > bv) { bv = probs[e]; best = e; }
            flat_e[t * KTOP + k] = best;
            flat_w[t * KTOP + k] = bv * inv;
            probs[best] = -2.f;
        }
    }
}

// ---------------- kernel 3: ordered capacity positions (single wave32) ----------------
__global__ __launch_bounds__(32) void pos_kernel(
    const int* __restrict__ flat_e, int* __restrict__ pos_out)
{
    __shared__ int cnt[NEXP];
    const int lane = threadIdx.x;
    for (int i = lane; i < NEXP; i += 32) cnt[i] = 0;
    __syncthreads();
    for (int base = 0; base < NFLAT; base += 32) {
        int e = flat_e[base + lane];
        int prior = 0, later = 0;
        for (int j = 0; j < 32; ++j) {
            int ej = __shfl(e, j, 32);
            if (ej == e) { if (j < lane) prior++; else if (j > lane) later++; }
        }
        int p = cnt[e] + prior;          // lockstep wave32: all reads precede write
        __syncthreads();
        if (later == 0) cnt[e] = p + 1;  // last matching lane commits the new count
        __syncthreads();
        pos_out[base + lane] = p;
    }
}

// ---------------- kernel 4: dispatch tokens into expert buffers ----------------
__global__ __launch_bounds__(256) void dispatch_kernel(
    const unsigned short* __restrict__ xnb, const int* __restrict__ flat_e,
    const int* __restrict__ pos, unsigned short* __restrict__ xe)
{
    const int i = blockIdx.x;
    const int p = pos[i];
    if (p >= CAP) return;
    const int e = flat_e[i];
    const int tok = i / KTOP;
    const unsigned short* src = xnb + (long)tok * HDIM;
    unsigned short* dst = xe + ((long)e * CAP + p) * HDIM;
    for (int h = threadIdx.x; h < HDIM / 2; h += blockDim.x)
        ((unsigned*)dst)[h] = ((const unsigned*)src)[h];
}

// ---------------- GEMM tiles ----------------
#define TM 64
#define TN 128
#define KS 32
#define APAD 40
#define BPAD 40

// C(bf16) = silu(A@B0) * (A@B1);  A bf16 [M,lda], B f32 [K,ldb]; batched on z
__global__ __launch_bounds__(256) void gemm_dual_silu(
    const unsigned short* __restrict__ Ab, long sA, int lda,
    const float* __restrict__ B0b, const float* __restrict__ B1b, long sB, int ldb,
    unsigned short* __restrict__ Cb, long sC, int ldc, int ksteps)
{
    __shared__ unsigned short Al[2][TM * APAD];
    __shared__ unsigned short Bl0[2][TN * BPAD];
    __shared__ unsigned short Bl1[2][TN * BPAD];

    const int bz = blockIdx.z;
    const int m0 = blockIdx.y * TM;
    const int n0 = blockIdx.x * TN;
    const unsigned short* A = Ab + (long)bz * sA;
    const float* B0 = B0b + (long)bz * sB;
    const float* B1 = B1b + (long)bz * sB;
    unsigned short* C = Cb + (long)bz * sC;

    const int tid  = threadIdx.x;
    const int lane = tid & 31;
    const int wave = tid >> 5;
    const int wm = wave & 1;    // 0..1 -> 32-row slab
    const int wn = wave >> 1;   // 0..3 -> 32-col slab
    const int hl = lane >> 4;   // lane half
    const int r  = lane & 15;

    // staging geometry
    const int arow = tid >> 2;         // 0..63
    const int acol = (tid & 3) * 8;    // bf16 element offset (16B chunks)
    const int quad = tid & 31;         // float4 column in N
    const int kp0  = tid >> 5;         // k-pair base 0..7

    const unsigned short* agp = A + (long)(m0 + arow) * lda + acol;  // + k0
    const unsigned a_lds0 = lds_off32(&Al[0][arow * APAD + acol]);
    const unsigned a_lds1 = lds_off32(&Al[1][arow * APAD + acol]);

    f32x8 zero = {0.f,0.f,0.f,0.f,0.f,0.f,0.f,0.f};
    f32x8 acc[2][2][2];
#pragma unroll
    for (int mi = 0; mi < 2; ++mi)
#pragma unroll
        for (int ni = 0; ni < 2; ++ni) { acc[mi][ni][0] = zero; acc[mi][ni][1] = zero; }

    float4 pb0[2][2], pb1[2][2];

    // ---- tile 0: async A -> buf0, B via regs+convert -> buf0 ----
    async_g2l_b128(a_lds0, agp);
#pragma unroll
    for (int it = 0; it < 2; ++it) {
        const int kk = 2 * (kp0 + it * 8);
        pb0[it][0] = *(const float4*)(B0 + (long)(kk    ) * ldb + n0 + quad * 4);
        pb0[it][1] = *(const float4*)(B0 + (long)(kk + 1) * ldb + n0 + quad * 4);
        pb1[it][0] = *(const float4*)(B1 + (long)(kk    ) * ldb + n0 + quad * 4);
        pb1[it][1] = *(const float4*)(B1 + (long)(kk + 1) * ldb + n0 + quad * 4);
    }
#pragma unroll
    for (int it = 0; it < 2; ++it) {
        const int kk = 2 * (kp0 + it * 8);
        const int nb = quad * 4;
        *(unsigned*)(&Bl0[0][(nb + 0) * BPAD + kk]) = cvt2(pb0[it][0].x, pb0[it][1].x);
        *(unsigned*)(&Bl0[0][(nb + 1) * BPAD + kk]) = cvt2(pb0[it][0].y, pb0[it][1].y);
        *(unsigned*)(&Bl0[0][(nb + 2) * BPAD + kk]) = cvt2(pb0[it][0].z, pb0[it][1].z);
        *(unsigned*)(&Bl0[0][(nb + 3) * BPAD + kk]) = cvt2(pb0[it][0].w, pb0[it][1].w);
        *(unsigned*)(&Bl1[0][(nb + 0) * BPAD + kk]) = cvt2(pb1[it][0].x, pb1[it][1].x);
        *(unsigned*)(&Bl1[0][(nb + 1) * BPAD + kk]) = cvt2(pb1[it][0].y, pb1[it][1].y);
        *(unsigned*)(&Bl1[0][(nb + 2) * BPAD + kk]) = cvt2(pb1[it][0].z, pb1[it][1].z);
        *(unsigned*)(&Bl1[0][(nb + 3) * BPAD + kk]) = cvt2(pb1[it][0].w, pb1[it][1].w);
    }
    wait_async0();
    __syncthreads();

    for (int ks = 0; ks < ksteps; ++ks) {
        const int buf  = ks & 1;
        const bool more = (ks + 1) < ksteps;

        // ---- prefetch next K-tile: A async into back buffer, B into regs ----
        if (more) {
            const int k0n = (ks + 1) * KS;
            async_g2l_b128(buf ? a_lds0 : a_lds1, agp + k0n);
#pragma unroll
            for (int it = 0; it < 2; ++it) {
                const int kk = k0n + 2 * (kp0 + it * 8);
                pb0[it][0] = *(const float4*)(B0 + (long)(kk    ) * ldb + n0 + quad * 4);
                pb0[it][1] = *(const float4*)(B0 + (long)(kk + 1) * ldb + n0 + quad * 4);
                pb1[it][0] = *(const float4*)(B1 + (long)(kk    ) * ldb + n0 + quad * 4);
                pb1[it][1] = *(const float4*)(B1 + (long)(kk + 1) * ldb + n0 + quad * 4);
            }
        }

        // ---- WMMA on current buffer ----
        FragCvt af[2], bf0[2], bf1[2];
#pragma unroll
        for (int mi = 0; mi < 2; ++mi) {
            const unsigned short* base = &Al[buf][(wm * 32 + mi * 16 + r) * APAD];
            // A layout: half0 -> K 0..15 ; half1 -> K 16..31 (contiguous)
            af[mi].u.a = *(const uint4*)(base + hl * 16);
            af[mi].u.b = *(const uint4*)(base + hl * 16 + 8);
        }
#pragma unroll
        for (int ni = 0; ni < 2; ++ni) {
            const unsigned short* p0 = &Bl0[buf][(wn * 32 + ni * 16 + r) * BPAD + hl * 16];
            const unsigned short* p1 = &Bl1[buf][(wn * 32 + ni * 16 + r) * BPAD + hl * 16];
            // B layout: half0 -> K 0..15 ; half1 -> K 16..31 (contiguous)
            bf0[ni].u.a = *(const uint4*)(p0);
            bf0[ni].u.b = *(const uint4*)(p0 + 8);
            bf1[ni].u.a = *(const uint4*)(p1);
            bf1[ni].u.b = *(const uint4*)(p1 + 8);
        }
#pragma unroll
        for (int mi = 0; mi < 2; ++mi)
#pragma unroll
            for (int ni = 0; ni < 2; ++ni) {
                acc[mi][ni][0] = wmma_bf16(af[mi].v, bf0[ni].v, acc[mi][ni][0]);
                acc[mi][ni][1] = wmma_bf16(af[mi].v, bf1[ni].v, acc[mi][ni][1]);
            }

        // ---- stage converted B tiles into the back buffer ----
        if (more) {
            const int nb = quad * 4;
#pragma unroll
            for (int it = 0; it < 2; ++it) {
                const int kk = 2 * (kp0 + it * 8);
                *(unsigned*)(&Bl0[buf ^ 1][(nb + 0) * BPAD + kk]) = cvt2(pb0[it][0].x, pb0[it][1].x);
                *(unsigned*)(&Bl0[buf ^ 1][(nb + 1) * BPAD + kk]) = cvt2(pb0[it][0].y, pb0[it][1].y);
                *(unsigned*)(&Bl0[buf ^ 1][(nb + 2) * BPAD + kk]) = cvt2(pb0[it][0].z, pb0[it][1].z);
                *(unsigned*)(&Bl0[buf ^ 1][(nb + 3) * BPAD + kk]) = cvt2(pb0[it][0].w, pb0[it][1].w);
                *(unsigned*)(&Bl1[buf ^ 1][(nb + 0) * BPAD + kk]) = cvt2(pb1[it][0].x, pb1[it][1].x);
                *(unsigned*)(&Bl1[buf ^ 1][(nb + 1) * BPAD + kk]) = cvt2(pb1[it][0].y, pb1[it][1].y);
                *(unsigned*)(&Bl1[buf ^ 1][(nb + 2) * BPAD + kk]) = cvt2(pb1[it][0].z, pb1[it][1].z);
                *(unsigned*)(&Bl1[buf ^ 1][(nb + 3) * BPAD + kk]) = cvt2(pb1[it][0].w, pb1[it][1].w);
            }
        }
        wait_async0();
        __syncthreads();
    }

    // fused SwiGLU epilogue (fully unrolled)
#pragma unroll
    for (int mi = 0; mi < 2; ++mi)
#pragma unroll
        for (int ni = 0; ni < 2; ++ni)
#pragma unroll
            for (int rr = 0; rr < 8; ++rr) {
                const float gv = acc[mi][ni][0][rr];
                const float uv = acc[mi][ni][1][rr];
                const float hv = (gv / (1.f + __expf(-gv))) * uv;
                const int row = m0 + wm * 32 + mi * 16 + hl * 8 + rr;
                const int col = n0 + wn * 32 + ni * 16 + r;
                C[(long)row * ldc + col] = f2bf(hv);
            }
}

// C(f32) = A@B;  A bf16 [M,lda], B f32 [K,ldb]; batched on z
__global__ __launch_bounds__(256) void gemm_single(
    const unsigned short* __restrict__ Ab, long sA, int lda,
    const float* __restrict__ Bb, long sB, int ldb,
    float* __restrict__ Cb, long sC, int ldc, int ksteps)
{
    __shared__ unsigned short Al[2][TM * APAD];
    __shared__ unsigned short Bl[2][TN * BPAD];

    const int bz = blockIdx.z;
    const int m0 = blockIdx.y * TM;
    const int n0 = blockIdx.x * TN;
    const unsigned short* A = Ab + (long)bz * sA;
    const float* B = Bb + (long)bz * sB;
    float* C = Cb + (long)bz * sC;

    const int tid  = threadIdx.x;
    const int lane = tid & 31;
    const int wave = tid >> 5;
    const int wm = wave & 1;
    const int wn = wave >> 1;
    const int hl = lane >> 4;
    const int r  = lane & 15;

    const int arow = tid >> 2;
    const int acol = (tid & 3) * 8;
    const int quad = tid & 31;
    const int kp0  = tid >> 5;

    const unsigned short* agp = A + (long)(m0 + arow) * lda + acol;
    const unsigned a_lds0 = lds_off32(&Al[0][arow * APAD + acol]);
    const unsigned a_lds1 = lds_off32(&Al[1][arow * APAD + acol]);

    f32x8 zero = {0.f,0.f,0.f,0.f,0.f,0.f,0.f,0.f};
    f32x8 acc[2][2];
#pragma unroll
    for (int mi = 0; mi < 2; ++mi)
#pragma unroll
        for (int ni = 0; ni < 2; ++ni) acc[mi][ni] = zero;

    float4 pb[2][2];

    async_g2l_b128(a_lds0, agp);
#pragma unroll
    for (int it = 0; it < 2; ++it) {
        const int kk = 2 * (kp0 + it * 8);
        pb[it][0] = *(const float4*)(B + (long)(kk    ) * ldb + n0 + quad * 4);
        pb[it][1] = *(const float4*)(B + (long)(kk + 1) * ldb + n0 + quad * 4);
    }
#pragma unroll
    for (int it = 0; it < 2; ++it) {
        const int kk = 2 * (kp0 + it * 8);
        const int nb = quad * 4;
        *(unsigned*)(&Bl[0][(nb + 0) * BPAD + kk]) = cvt2(pb[it][0].x, pb[it][1].x);
        *(unsigned*)(&Bl[0][(nb + 1) * BPAD + kk]) = cvt2(pb[it][0].y, pb[it][1].y);
        *(unsigned*)(&Bl[0][(nb + 2) * BPAD + kk]) = cvt2(pb[it][0].z, pb[it][1].z);
        *(unsigned*)(&Bl[0][(nb + 3) * BPAD + kk]) = cvt2(pb[it][0].w, pb[it][1].w);
    }
    wait_async0();
    __syncthreads();

    for (int ks = 0; ks < ksteps; ++ks) {
        const int buf  = ks & 1;
        const bool more = (ks + 1) < ksteps;

        if (more) {
            const int k0n = (ks + 1) * KS;
            async_g2l_b128(buf ? a_lds0 : a_lds1, agp + k0n);
#pragma unroll
            for (int it = 0; it < 2; ++it) {
                const int kk = k0n + 2 * (kp0 + it * 8);
                pb[it][0] = *(const float4*)(B + (long)(kk    ) * ldb + n0 + quad * 4);
                pb[it][1] = *(const float4*)(B + (long)(kk + 1) * ldb + n0 + quad * 4);
            }
        }

        FragCvt af[2], bfr[2];
#pragma unroll
        for (int mi = 0; mi < 2; ++mi) {
            const unsigned short* base = &Al[buf][(wm * 32 + mi * 16 + r) * APAD];
            af[mi].u.a = *(const uint4*)(base + hl * 16);
            af[mi].u.b = *(const uint4*)(base + hl * 16 + 8);
        }
#pragma unroll
        for (int ni = 0; ni < 2; ++ni) {
            const unsigned short* p0 = &Bl[buf][(wn * 32 + ni * 16 + r) * BPAD + hl * 16];
            bfr[ni].u.a = *(const uint4*)(p0);
            bfr[ni].u.b = *(const uint4*)(p0 + 8);
        }
#pragma unroll
        for (int mi = 0; mi < 2; ++mi)
#pragma unroll
            for (int ni = 0; ni < 2; ++ni)
                acc[mi][ni] = wmma_bf16(af[mi].v, bfr[ni].v, acc[mi][ni]);

        if (more) {
            const int nb = quad * 4;
#pragma unroll
            for (int it = 0; it < 2; ++it) {
                const int kk = 2 * (kp0 + it * 8);
                *(unsigned*)(&Bl[buf ^ 1][(nb + 0) * BPAD + kk]) = cvt2(pb[it][0].x, pb[it][1].x);
                *(unsigned*)(&Bl[buf ^ 1][(nb + 1) * BPAD + kk]) = cvt2(pb[it][0].y, pb[it][1].y);
                *(unsigned*)(&Bl[buf ^ 1][(nb + 2) * BPAD + kk]) = cvt2(pb[it][0].z, pb[it][1].z);
                *(unsigned*)(&Bl[buf ^ 1][(nb + 3) * BPAD + kk]) = cvt2(pb[it][0].w, pb[it][1].w);
            }
        }
        wait_async0();
        __syncthreads();
    }

#pragma unroll
    for (int mi = 0; mi < 2; ++mi)
#pragma unroll
        for (int ni = 0; ni < 2; ++ni)
#pragma unroll
            for (int rr = 0; rr < 8; ++rr) {
                const int row = m0 + wm * 32 + mi * 16 + hl * 8 + rr;
                const int col = n0 + wn * 32 + ni * 16 + r;
                C[(long)row * ldc + col] = acc[mi][ni][rr];
            }
}

// ---------------- kernel 9: combine ----------------
__global__ __launch_bounds__(256) void combine_kernel(
    const float* __restrict__ hidden, const float* __restrict__ shout,
    const float* __restrict__ ye, const int* __restrict__ flat_e,
    const int* __restrict__ pos, const float* __restrict__ flat_w,
    float* __restrict__ out)
{
    const int t = blockIdx.x;
    int   ee[KTOP], pp[KTOP];
    float ww[KTOP];
#pragma unroll
    for (int k = 0; k < KTOP; ++k) {
        ee[k] = flat_e[t * KTOP + k];
        pp[k] = pos[t * KTOP + k];
        ww[k] = flat_w[t * KTOP + k];
    }
    for (int h = threadIdx.x; h < HDIM; h += 256) {
        float acc = hidden[(long)t * HDIM + h] + shout[(long)t * HDIM + h];
#pragma unroll
        for (int k = 0; k < KTOP; ++k)
            if (pp[k] < CAP)
                acc = fmaf(ww[k], ye[((long)ee[k] * CAP + pp[k]) * HDIM + h], acc);
        out[(long)t * HDIM + h] = acc;
    }
}

// ---------------- launch ----------------
static inline size_t alignup(size_t x) { return (x + 255) & ~(size_t)255; }

extern "C" void kernel_launch(void* const* d_in, const int* in_sizes, int n_in,
                              void* d_out, int out_size, void* d_ws, size_t ws_size,
                              hipStream_t stream) {
    const float* hidden   = (const float*)d_in[0];
    const float* ln_w     = (const float*)d_in[3];
    const float* router_w = (const float*)d_in[4];
    const float* wg       = (const float*)d_in[5];
    const float* wu       = (const float*)d_in[6];
    const float* wd       = (const float*)d_in[7];
    const float* ws_g     = (const float*)d_in[8];
    const float* ws_u     = (const float*)d_in[9];
    const float* ws_d     = (const float*)d_in[10];
    float* out = (float*)d_out;

    char* p = (char*)d_ws;
    size_t off = 0;
    float*          xn   = (float*)         (p + off); off = alignup(off + (size_t)T_TOK * HDIM * 4);
    unsigned short* xnb  = (unsigned short*)(p + off); off = alignup(off + (size_t)T_TOK * HDIM * 2);
    int*            fe   = (int*)           (p + off); off = alignup(off + (size_t)NFLAT * 4);
    float*          fw   = (float*)         (p + off); off = alignup(off + (size_t)NFLAT * 4);
    int*            fpos = (int*)           (p + off); off = alignup(off + (size_t)NFLAT * 4);
    unsigned short* xe   = (unsigned short*)(p + off); off = alignup(off + (size_t)NEXP * CAP * HDIM * 2);
    unsigned short* hbuf = (unsigned short*)(p + off); off = alignup(off + (size_t)NEXP * CAP * FDIM * 2);
    float*          ye   = (float*)         (p + off); off = alignup(off + (size_t)NEXP * CAP * HDIM * 4);
    unsigned short* hsh  = (unsigned short*)(p + off); off = alignup(off + (size_t)T_TOK * FSDIM * 2);
    float*          shout= (float*)         (p + off); off = alignup(off + (size_t)T_TOK * HDIM * 4);

    rmsnorm_kernel<<<T_TOK, 256, 0, stream>>>(hidden, ln_w, xn, xnb);
    router_kernel<<<T_TOK, 64, 0, stream>>>(xn, router_w, fe, fw);
    pos_kernel<<<1, 32, 0, stream>>>(fe, fpos);
    dispatch_kernel<<<NFLAT, 256, 0, stream>>>(xnb, fe, fpos, xe);

    // routed experts: gate/up + SwiGLU  [64] x (192x896 = 192x1280 @ 1280x896)
    gemm_dual_silu<<<dim3(FDIM / TN, CAP / TM, NEXP), 256, 0, stream>>>(
        xe, (long)CAP * HDIM, HDIM,
        wg, wu, (long)HDIM * FDIM, FDIM,
        hbuf, (long)CAP * FDIM, FDIM, HDIM / KS);

    // shared expert: gate/up + SwiGLU  (1024x1792 = 1024x1280 @ 1280x1792)
    gemm_dual_silu<<<dim3(FSDIM / TN, T_TOK / TM, 1), 256, 0, stream>>>(
        xnb, 0, HDIM,
        ws_g, ws_u, 0, FSDIM,
        hsh, 0, FSDIM, HDIM / KS);

    // routed experts: down  [64] x (192x1280 = 192x896 @ 896x1280)
    gemm_single<<<dim3(HDIM / TN, CAP / TM, NEXP), 256, 0, stream>>>(
        hbuf, (long)CAP * FDIM, FDIM,
        wd, (long)FDIM * HDIM, HDIM,
        ye, (long)CAP * HDIM, HDIM, FDIM / KS);

    // shared expert: down  (1024x1280 = 1024x1792 @ 1792x1280)
    gemm_single<<<dim3(HDIM / TN, T_TOK / TM, 1), 256, 0, stream>>>(
        hsh, 0, FSDIM,
        ws_d, 0, HDIM,
        shout, 0, HDIM, FSDIM / KS);

    combine_kernel<<<T_TOK, 256, 0, stream>>>(hidden, shout, ye, fe, fpos, fw, out);
}
